// HeposMultiHeadAttention_33715493273892
// MI455X (gfx1250) — compile-verified, matrix operand order checked by reading, not run on hardware
//
#include <hip/hip_runtime.h>
#include <hip/hip_bf16.h>
#include <math.h>

typedef _Float16 v16h  __attribute__((ext_vector_type(16)));
typedef _Float16 half8 __attribute__((ext_vector_type(8)));
typedef _Float16 half4 __attribute__((ext_vector_type(4)));
typedef float    v8f   __attribute__((ext_vector_type(8)));

#define WMMA_F32_F16(a, b, c) \
  __builtin_amdgcn_wmma_f32_16x16x32_f16(false, (a), false, (b), (short)0, (c), false, false)

static constexpr int Bc = 4, TQc = 2048, TKc = 8192, Dc = 1024;
static constexpr int Hc = 16, DHc = 64, Lc = 512;

// Load a 16x32 f16 WMMA A/B fragment from an LDS (or global) tile.
// A-layout per CDNA5 ISA: lane L holds row (row_base + L%16); halves j cover
// k = kb + {0..7} and k = kb + {16..23}, kb = 8*(L>=16) + koff.
__device__ __forceinline__ v16h ldfrag(const _Float16* t, int stride, int row_base,
                                       int koff, int lane) {
  int r  = row_base + (lane & 15);
  int kb = ((lane >> 4) & 1) * 8 + koff;
  half8 lo = *(const half8*)(t + r * stride + kb);
  half8 hi = *(const half8*)(t + r * stride + kb + 16);
  return __builtin_shufflevector(lo, hi, 0, 1, 2, 3, 4, 5, 6, 7,
                                 8, 9, 10, 11, 12, 13, 14, 15);
}

// Async memory->LDS 128-bit copy (CDNA5; tracked by ASYNCcnt).
__device__ __forceinline__ void async_b128(const _Float16* g, _Float16* l) {
  unsigned loff = (unsigned)(unsigned long long)(uintptr_t)l;  // low 32b = LDS offset
  unsigned long long ga = (unsigned long long)(uintptr_t)g;
  asm volatile("global_load_async_to_lds_b128 %0, %1, off" :: "v"(loff), "v"(ga) : "memory");
}
__device__ __forceinline__ void wait_async0() {
  asm volatile("s_wait_asynccnt 0x0" ::: "memory");
}

// ---------------------------------------------------------------------------
// Projection GEMM: Y_f16 = (X_f32 @ W_f32^T + bias) * scale
// MODE 0: Q.  grid (TQ/64, H, B); out Qh[((b*H+h)*TQ + t)*64 + d]   (row major)
// MODE 1: K (HEPOS gather rows t=h+16*l). out Kg[((b*H+h)*L + l)*64 + d]
// MODE 2: V (HEPOS gather) TRANSPOSED out: VgT[((b*H+h)*64 + d)*L + l]
// 64x64 tile / block, 4 waves each computing a 32x32 quadrant via 2x2 WMMAs.
// ---------------------------------------------------------------------------
template <int MODE>
__global__ __launch_bounds__(128) void proj_kernel(
    const float* __restrict__ X, const float* __restrict__ W,
    const float* __restrict__ bias, _Float16* __restrict__ Y, float scale) {
  __shared__ __align__(16) _Float16 At[64 * 40];
  __shared__ __align__(16) _Float16 Bt[64 * 40];
  const int tid = threadIdx.x, lane = tid & 31;
  const int w = tid >> 5, wr = w >> 1, wc = w & 1;
  const int tile = blockIdx.x, h = blockIdx.y, b = blockIdx.z;
  const int n0 = h * DHc;

  v8f acc[2][2] = {};
  for (int k0 = 0; k0 < Dc; k0 += 32) {
    __syncthreads();
#pragma unroll
    for (int e = 0; e < 4; ++e) {  // stage A 64x32: float4 load -> half4 store
      int idx = (e * 128 + tid) * 4;
      int r = idx >> 5, c = idx & 31;
      long gr = (MODE == 0) ? ((long)b * TQc + tile * 64 + r)
                            : ((long)b * TKc + h + 16 * (tile * 64 + r));
      float4 f = *(const float4*)(&X[gr * Dc + k0 + c]);
      half4 hh = {(_Float16)f.x, (_Float16)f.y, (_Float16)f.z, (_Float16)f.w};
      *(half4*)(At + r * 40 + c) = hh;
    }
#pragma unroll
    for (int e = 0; e < 4; ++e) {  // stage B = W rows n0..n0+63
      int idx = (e * 128 + tid) * 4;
      int r = idx >> 5, c = idx & 31;
      float4 f = *(const float4*)(&W[(long)(n0 + r) * Dc + k0 + c]);
      half4 hh = {(_Float16)f.x, (_Float16)f.y, (_Float16)f.z, (_Float16)f.w};
      *(half4*)(Bt + r * 40 + c) = hh;
    }
    __syncthreads();
    v16h a0 = ldfrag(At, 40, wr * 32, 0, lane);
    v16h a1 = ldfrag(At, 40, wr * 32 + 16, 0, lane);
    v16h b0 = ldfrag(Bt, 40, wc * 32, 0, lane);
    v16h b1 = ldfrag(Bt, 40, wc * 32 + 16, 0, lane);
    acc[0][0] = WMMA_F32_F16(a0, b0, acc[0][0]);
    acc[0][1] = WMMA_F32_F16(a0, b1, acc[0][1]);
    acc[1][0] = WMMA_F32_F16(a1, b0, acc[1][0]);
    acc[1][1] = WMMA_F32_F16(a1, b1, acc[1][1]);
  }
  const int base8 = ((lane >> 4) & 1) * 8;
#pragma unroll
  for (int i = 0; i < 2; ++i)
#pragma unroll
    for (int j = 0; j < 2; ++j) {
      int nl = wc * 32 + j * 16 + (lane & 15);
      float bv = bias[n0 + nl];
      if (MODE == 2) {  // transposed: row d=nl, 8 consecutive l -> one b128
        int l = tile * 64 + wr * 32 + i * 16 + base8;
        half8 pk;
#pragma unroll
        for (int v = 0; v < 8; ++v) pk[v] = (_Float16)((acc[i][j][v] + bv) * scale);
        *(half8*)(Y + (((long)b * Hc + h) * DHc + nl) * Lc + l) = pk;
      } else {
#pragma unroll
        for (int v = 0; v < 8; ++v) {
          int ml = wr * 32 + i * 16 + base8 + v;
          float y = (acc[i][j][v] + bv) * scale;
          long o = (MODE == 0)
                       ? ((((long)b * Hc + h) * TQc + (tile * 64 + ml)) * DHc + nl)
                       : ((((long)b * Hc + h) * Lc + (tile * 64 + ml)) * DHc + nl);
          Y[o] = (_Float16)y;
        }
      }
    }
}

// ---------------------------------------------------------------------------
// Flash attention over gathered keys. grid (TQ/64, H, B), 4 waves/block,
// each wave owns 16 query rows. 64-key chunks (16 WMMA per chunk, softmax
// reduction cost amortized 2x vs 32-key chunks). K chunk staged [64 l x 64 d],
// V chunk pre-transposed [64 d x 64 l]. Double-buffered async staging:
// chunk i+1's copies overlap chunk i's compute.
// ---------------------------------------------------------------------------
__global__ __launch_bounds__(128) void attn_kernel(
    const _Float16* __restrict__ Qh, const _Float16* __restrict__ Kg,
    const _Float16* __restrict__ VgT, _Float16* __restrict__ O) {
  __shared__ __align__(16) _Float16 kt[2][64 * 72];
  __shared__ __align__(16) _Float16 vt[2][64 * 72];
  __shared__ __align__(16) _Float16 pt[4][16 * 72];
  const int tid = threadIdx.x, lane = tid & 31, w = tid >> 5;
  const int qt = blockIdx.x, h = blockIdx.y, b = blockIdx.z;
  const int q0 = qt * 64 + w * 16;

  const _Float16* Qb = Qh + (((long)b * Hc + h) * TQc + q0) * DHc;
  const _Float16* Kb = Kg + (((long)b * Hc + h) * Lc) * DHc;
  const _Float16* Vb = VgT + (((long)b * Hc + h) * DHc) * Lc;

  v16h qa0 = ldfrag(Qb, DHc, 0, 0, lane);   // d = 0..31
  v16h qa1 = ldfrag(Qb, DHc, 0, 32, lane);  // d = 32..63

  float mrun[8], lrun[8];
  v8f oacc[4] = {};
#pragma unroll
  for (int j = 0; j < 8; ++j) { mrun[j] = -1e30f; lrun[j] = 0.f; }
  _Float16* P = &pt[w][0];
  const int col = lane & 15, rb = ((lane >> 4) & 1) * 8;

  // staging coordinates: each thread copies 32 halves of K and of V^T
  const int sr = tid >> 1, sc = (tid & 1) * 32;

  // prologue: stage chunk 0 into buffer 0
#pragma unroll
  for (int u = 0; u < 4; ++u) {
    async_b128(Kb + (long)sr * DHc + sc + u * 8, &kt[0][sr * 72 + sc + u * 8]);
    async_b128(Vb + (long)sr * Lc + sc + u * 8, &vt[0][sr * 72 + sc + u * 8]);
  }
  wait_async0();
  __syncthreads();

  const int NCH = Lc / 64;  // 8 chunks
  for (int c = 0; c < NCH; ++c) {
    const int cur = c & 1;
    if (c + 1 < NCH) {  // issue next chunk's copies; they overlap compute below
      int l1 = (c + 1) * 64;
#pragma unroll
      for (int u = 0; u < 4; ++u) {
        async_b128(Kb + (long)(l1 + sr) * DHc + sc + u * 8,
                   &kt[cur ^ 1][sr * 72 + sc + u * 8]);
        async_b128(Vb + (long)sr * Lc + l1 + sc + u * 8,
                   &vt[cur ^ 1][sr * 72 + sc + u * 8]);
      }
    }
    const _Float16* K = &kt[cur][0];
    const _Float16* V = &vt[cur][0];

    v8f z = {};
    v8f s[4];
#pragma unroll
    for (int t = 0; t < 4; ++t) {  // scores: 16 q x 64 l
      s[t] = WMMA_F32_F16(qa0, ldfrag(K, 72, t * 16, 0, lane), z);
      s[t] = WMMA_F32_F16(qa1, ldfrag(K, 72, t * 16, 32, lane), s[t]);
    }

    // Online softmax. C-layout: vgpr j holds row rb+j, lane%16 = col.
    // Row-max reduced per chunk; row-sum stays lane-partial until the end.
#pragma unroll
    for (int j = 0; j < 8; ++j) {
      float cmax = fmaxf(fmaxf(s[0][j], s[1][j]), fmaxf(s[2][j], s[3][j]));
#pragma unroll
      for (int mm = 1; mm < 16; mm <<= 1) cmax = fmaxf(cmax, __shfl_xor(cmax, mm, 32));
      float nm = fmaxf(mrun[j], cmax);
      float corr = __expf(mrun[j] - nm);
      mrun[j] = nm;
      float es = 0.f;
#pragma unroll
      for (int t = 0; t < 4; ++t) {
        float e = __expf(s[t][j] - nm);
        es += e;
        P[(rb + j) * 72 + t * 16 + col] = (_Float16)e;  // straight to LDS
        oacc[t][j] *= corr;
      }
      lrun[j] = lrun[j] * corr + es;
    }
    asm volatile("s_wait_dscnt 0x0" ::: "memory");
    v16h pf0 = ldfrag(P, 72, 0, 0, lane);   // P, l = 0..31
    v16h pf1 = ldfrag(P, 72, 0, 32, lane);  // P, l = 32..63
#pragma unroll
    for (int t = 0; t < 4; ++t) {  // O += P(16x64) @ V(64x16) per d-tile
      oacc[t] = WMMA_F32_F16(pf0, ldfrag(V, 72, t * 16, 0, lane), oacc[t]);
      oacc[t] = WMMA_F32_F16(pf1, ldfrag(V, 72, t * 16, 32, lane), oacc[t]);
    }
    wait_async0();    // next chunk's async copies complete
    __syncthreads();  // ... and visible to all waves
  }

  // single deferred row-sum reduction, normalize, store f16 [B, TQ, D]
#pragma unroll
  for (int j = 0; j < 8; ++j) {
    float rs = lrun[j];
#pragma unroll
    for (int mm = 1; mm < 16; mm <<= 1) rs += __shfl_xor(rs, mm, 32);
    float inv = 1.f / rs;
    long base = ((long)b * TQc + (q0 + rb + j)) * Dc + h * DHc;
#pragma unroll
    for (int t = 0; t < 4; ++t)
      O[base + t * 16 + col] = (_Float16)(oacc[t][j] * inv);
  }
}

// ---------------------------------------------------------------------------
// Output projection: out_f32[M,1024] = O_f16[M,1024] @ Wo^T + bo
// grid ((B*TQ)/64, 1024/64). A-tile staged via async copies (pure f16 copy),
// overlapped with the f32->f16 W-tile conversion.
// ---------------------------------------------------------------------------
__global__ __launch_bounds__(128) void outproj_kernel(
    const _Float16* __restrict__ A, const float* __restrict__ W,
    const float* __restrict__ bias, float* __restrict__ Out) {
  __shared__ __align__(16) _Float16 At[64 * 40];
  __shared__ __align__(16) _Float16 Bt[64 * 40];
  const int tid = threadIdx.x, lane = tid & 31;
  const int w = tid >> 5, wr = w >> 1, wc = w & 1;
  const int m0 = blockIdx.x * 64, n0 = blockIdx.y * 64;

  v8f acc[2][2] = {};
  for (int k0 = 0; k0 < Dc; k0 += 32) {
    __syncthreads();
    {  // stage A (already f16): async 64x32 copy
      int r = tid >> 1, c = (tid & 1) * 16;
      const _Float16* as = A + (long)(m0 + r) * Dc + k0 + c;
      async_b128(as, At + r * 40 + c);
      async_b128(as + 8, At + r * 40 + c + 8);
    }
#pragma unroll
    for (int e = 0; e < 4; ++e) {  // stage B (f32 -> f16) vectorized, overlaps A
      int idx = (e * 128 + tid) * 4;
      int r = idx >> 5, c = idx & 31;
      float4 f = *(const float4*)(&W[(long)(n0 + r) * Dc + k0 + c]);
      half4 hh = {(_Float16)f.x, (_Float16)f.y, (_Float16)f.z, (_Float16)f.w};
      *(half4*)(Bt + r * 40 + c) = hh;
    }
    wait_async0();
    __syncthreads();
    v16h a0 = ldfrag(At, 40, wr * 32, 0, lane);
    v16h a1 = ldfrag(At, 40, wr * 32 + 16, 0, lane);
    v16h b0 = ldfrag(Bt, 40, wc * 32, 0, lane);
    v16h b1 = ldfrag(Bt, 40, wc * 32 + 16, 0, lane);
    acc[0][0] = WMMA_F32_F16(a0, b0, acc[0][0]);
    acc[0][1] = WMMA_F32_F16(a0, b1, acc[0][1]);
    acc[1][0] = WMMA_F32_F16(a1, b0, acc[1][0]);
    acc[1][1] = WMMA_F32_F16(a1, b1, acc[1][1]);
  }
#pragma unroll
  for (int i = 0; i < 2; ++i)
#pragma unroll
    for (int j = 0; j < 2; ++j) {
      int nl = wc * 32 + j * 16 + (lane & 15);
      float bv = bias[n0 + nl];
#pragma unroll
      for (int v = 0; v < 8; ++v) {
        int ml = wr * 32 + i * 16 + ((lane >> 4) & 1) * 8 + v;
        Out[(long)(m0 + ml) * Dc + n0 + nl] = acc[i][j][v] + bv;
      }
    }
}

extern "C" void kernel_launch(void* const* d_in, const int* in_sizes, int n_in,
                              void* d_out, int out_size, void* d_ws, size_t ws_size,
                              hipStream_t stream) {
  const float* q  = (const float*)d_in[0];
  const float* k  = (const float*)d_in[1];
  const float* v  = (const float*)d_in[2];
  const float* Wq = (const float*)d_in[3];
  const float* bq = (const float*)d_in[4];
  const float* Wk = (const float*)d_in[5];
  const float* bk = (const float*)d_in[6];
  const float* Wv = (const float*)d_in[7];
  const float* bv = (const float*)d_in[8];
  const float* Wo = (const float*)d_in[9];
  const float* bo = (const float*)d_in[10];
  float* out = (float*)d_out;

  char* ws = (char*)d_ws;
  _Float16* Qh  = (_Float16*)ws;  // [B,H,TQ,64] f16
  _Float16* Kg  = (_Float16*)(ws + (size_t)Bc * Hc * TQc * DHc * 2);
  _Float16* VgT = (_Float16*)((char*)Kg + (size_t)Bc * Hc * Lc * DHc * 2);
  _Float16* Oa  = (_Float16*)((char*)VgT + (size_t)Bc * Hc * Lc * DHc * 2);  // [B,TQ,D] f16

  dim3 blk(128);
  // Q projection with 1/sqrt(dh) folded in
  proj_kernel<0><<<dim3(TQc / 64, Hc, Bc), blk, 0, stream>>>(q, Wq, bq, Qh, 0.125f);
  // HEPOS-gathered K/V projections (16x fewer FLOPs than dense K/V proj);
  // V written transposed [b,h,d,l] for contiguous P@V fragment loads.
  proj_kernel<1><<<dim3(Lc / 64, Hc, Bc), blk, 0, stream>>>(k, Wk, bk, Kg, 1.0f);
  proj_kernel<2><<<dim3(Lc / 64, Hc, Bc), blk, 0, stream>>>(v, Wv, bv, VgT, 1.0f);
  attn_kernel<<<dim3(TQc / 64, Hc, Bc), blk, 0, stream>>>(Qh, Kg, VgT, Oa);
  outproj_kernel<<<dim3((Bc * TQc) / 64, Dc / 64), blk, 0, stream>>>(Oa, Wo, bo, out);
}
